// MultiHeadedAttention_61435212202638
// MI455X (gfx1250) — matrix operand order repaired and checked by reading, not compiled, not run on hardware
//
#include <hip/hip_runtime.h>
#include <hip/hip_bf16.h>

// ---------------------------------------------------------------------------
// MultiHeadedAttention forward for MI455X (gfx1250), wave32 + WMMA f16/f32.
//  - Projection GEMMs: A tiles via per-lane async global->LDS (ASYNCcnt),
//    W tiles via Tensor Data Mover descriptors (TENSORcnt), double-buffered.
//  - Attention: fused scores+mask+softmax+PV per 16-row q-tile; score row
//    lives in LDS (f16), p_attn hits HBM exactly once (vectorized stores).
// ---------------------------------------------------------------------------

typedef _Float16 half8 __attribute__((ext_vector_type(8)));
typedef _Float16 v16h  __attribute__((ext_vector_type(16)));
typedef float    v8f   __attribute__((ext_vector_type(8)));
typedef float    f32x4 __attribute__((ext_vector_type(4)));
typedef unsigned int u32x4 __attribute__((ext_vector_type(4)));
typedef int          i32x8 __attribute__((ext_vector_type(8)));

#define B_  2
#define S_  2048
#define E_  1024
#define H_  16
#define D_  64

__device__ __forceinline__ v16h combine8(half8 lo, half8 hi) {
  v16h r;
#pragma unroll
  for (int i = 0; i < 8; i++) { r[i] = lo[i]; r[8 + i] = hi[i]; }
  return r;
}

// Load a 16-f16 fragment from two contiguous 16-byte chunks.
__device__ __forceinline__ v16h load_frag(const _Float16* lo, const _Float16* hi) {
  return combine8(*(const half8*)lo, *(const half8*)hi);
}

__device__ __forceinline__ v8f wmma_f16(v16h a, v16h b, v8f c) {
  return __builtin_amdgcn_wmma_f32_16x16x32_f16(false, a, false, b, (short)0, c,
                                                false, false);
}

// Wave-relative LDS byte offset of a __shared__ object: the flat aperture maps
// addr[31:0] directly to the LDS address, so truncation gives the DS offset.
__device__ __forceinline__ unsigned lds_off(const void* p) {
  return (unsigned)(unsigned long long)(uintptr_t)p;
}

// Async copy of 16 bytes global -> LDS (per-lane), tracked by ASYNCcnt.
__device__ __forceinline__ void async_copy_b128(const void* gsrc, void* ldst) {
  unsigned lofs = lds_off(ldst);
  unsigned long long ga = (unsigned long long)(uintptr_t)gsrc;
  asm volatile("global_load_async_to_lds_b128 %0, %1, off"
               :: "v"(lofs), "v"(ga)
               : "memory");
}

__device__ __forceinline__ void wait_async0() {
  asm volatile("s_wait_asynccnt 0x0" ::: "memory");
}

// Tensor Data Mover: load a 2-D f16 tile (tile_x elems contiguous per row,
// tile_y rows, row stride in elems) from global into LDS at lds_byte_off,
// padding pad_amount_code (+1 DWORDs) after every pad_interval DWORDs.
// D# per CDNA5 ISA ch.8: group0 = {count, lds_addr, global_addr, type=2},
// group1 = {data_size/pad flags, tensor dims, tile dims, strides}.
__device__ __forceinline__ void tdm_load_2d_f16(const void* gsrc,
                                                unsigned lds_byte_off,
                                                int tile_x, int tile_y,
                                                int stride_elems,
                                                int tensor_d0, int tensor_d1,
                                                int pad_en, int pad_interval_code,
                                                int pad_amount_code) {
  unsigned long long ga = (unsigned long long)(uintptr_t)gsrc;
  u32x4 g0;
  g0[0] = 1u;                                   // count=1, user descriptor
  g0[1] = lds_byte_off;                         // lds_addr (bytes)
  g0[2] = (unsigned)(ga & 0xFFFFFFFFull);       // global_addr lo
  g0[3] = (unsigned)(ga >> 32) | (2u << 30);    // global_addr hi | type=2
  i32x8 g1;
  g1[0] = (1 << 16) | (pad_en << 20) |          // data_size=1 (2 bytes), pad
          (pad_interval_code << 22) | (pad_amount_code << 25);
  g1[1] = (tensor_d0 & 0xFFFF) << 16;           // [47:32]=0, dim0 lo16
  g1[2] = ((tensor_d0 >> 16) & 0xFFFF) | ((tensor_d1 & 0xFFFF) << 16);
  g1[3] = ((tensor_d1 >> 16) & 0xFFFF) | ((tile_x & 0xFFFF) << 16);
  g1[4] = (tile_y & 0xFFFF);                    // tile_dim1 | tile_dim2=0
  g1[5] = stride_elems;                         // tensor_dim0_stride lo32
  g1[6] = 0;                                    // stride hi | dim1_stride lo
  g1[7] = 0;
  asm volatile("tensor_load_to_lds %0, %1" :: "s"(g0), "s"(g1) : "memory");
}

// ---------------------------------------------------------------------------
// f32 -> f16 pack kernel
// ---------------------------------------------------------------------------
__global__ void f32_to_f16_kernel(const float* __restrict__ src,
                                  _Float16* __restrict__ dst, int n) {
  int i = blockIdx.x * blockDim.x + threadIdx.x;
  int stride = gridDim.x * blockDim.x;
  for (; i < n; i += stride) dst[i] = (_Float16)src[i];
}

// ---------------------------------------------------------------------------
// Tiled WMMA GEMM:  C[4096][1024] = A[4096][1024] * W[1024][1024]^T + bias
//   OUTMODE 0: f16 out, head-split [b][h][s][d]
//   OUTMODE 1: f16 out, head-split transposed [b][h][d][s]   (for V)
//   OUTMODE 2: f32 out, plain [M][N]                         (final proj)
// 256 threads (8 waves) -> 128x128 tile; double-buffered K staging with
// per-lane async loads (A tile) + one TDM descriptor per K-step (W tile).
// ---------------------------------------------------------------------------
template <int OUTMODE>
__global__ __launch_bounds__(256)
void gemm128(const _Float16* __restrict__ A, const _Float16* __restrict__ W,
             const float* __restrict__ bias, _Float16* __restrict__ out16,
             float* __restrict__ out32) {
  __shared__ _Float16 As[2][128][40];  // 32 used + 8 pad (bank spread, 16B align)
  __shared__ _Float16 Ws[2][128][40];  // TDM pad: 4 DWORDs after every 16 DWORDs

  const int tid  = threadIdx.x;
  const int lane = tid & 31, wave = tid >> 5;
  const int hl   = lane >> 4, l15 = lane & 15;
  const int wm   = wave >> 2, wn = wave & 3;
  const int mbase = blockIdx.x * 128, nbase = blockIdx.y * 128;

  const v8f zero = {0, 0, 0, 0, 0, 0, 0, 0};
  v8f acc[4][2];
#pragma unroll
  for (int i = 0; i < 4; i++)
#pragma unroll
    for (int j = 0; j < 2; j++) acc[i][j] = zero;

  const int srow  = tid >> 1;  // 0..127
  const int shalf = tid & 1;   // which 16-element half of the 32-wide K tile
  const _Float16* gA = A + (size_t)(mbase + srow) * 1024 + shalf * 16;
  const _Float16* gWtile = W + (size_t)nbase * 1024;  // row 0 of W tile

  // Stage one 128x32 K-tile: A per-lane async; W via one TDM descriptor.
  auto stage = [&](int buf, int kk) {
    async_copy_b128(gA + kk,     &As[buf][srow][shalf * 16]);
    async_copy_b128(gA + kk + 8, &As[buf][srow][shalf * 16 + 8]);
    if (wave == 0) {
      // 32 elems/row (64B = 16 DWORDs) x 128 rows, stride 1024 elems;
      // pad 4 DWORDs (code 3) after every 16 DWORDs (code 3) -> [128][40].
      tdm_load_2d_f16(gWtile + kk, lds_off(&Ws[buf][0][0]),
                      /*tile_x=*/32, /*tile_y=*/128, /*stride=*/1024,
                      /*tensor_d0=*/1024, /*tensor_d1=*/1024,
                      /*pad_en=*/1, /*pad_interval_code=*/3, /*pad_amount_code=*/3);
    }
  };

  stage(0, 0);  // prologue

  for (int it = 0; it < 32; ++it) {
    const int kk  = it * 32;
    const int cur = it & 1;

    wait_async0();                        // my A-tile copies for buf[cur]
    __builtin_amdgcn_s_wait_tensorcnt(0); // wave0's W-tile TDM (no-op elsewhere)
    __syncthreads();                      // all writes to buf[cur] visible

    if (it + 1 < 32) stage(1 - cur, kk + 32);  // overlap next tile's fetch

    v16h bfrag[2];
#pragma unroll
    for (int nt = 0; nt < 2; nt++) {
      const _Float16* wr = &Ws[cur][wn * 32 + nt * 16 + l15][0];
      bfrag[nt] = load_frag(wr + hl * 8, wr + 16 + hl * 8);
    }
#pragma unroll
    for (int mt = 0; mt < 4; mt++) {
      const _Float16* ar = &As[cur][wm * 64 + mt * 16 + l15][0];
      v16h afrag = load_frag(ar + hl * 8, ar + 16 + hl * 8);
#pragma unroll
      for (int nt = 0; nt < 2; nt++) {
        acc[mt][nt] = wmma_f16(afrag, bfrag[nt], acc[mt][nt]);
      }
    }
  }

  // Epilogue. C layout: VGPR r -> M = hl*8 + r, N = l15.
#pragma unroll
  for (int mt = 0; mt < 4; mt++) {
#pragma unroll
    for (int nt = 0; nt < 2; nt++) {
#pragma unroll
      for (int r = 0; r < 8; r++) {
        const int gm = mbase + wm * 64 + mt * 16 + hl * 8 + r;
        const int gn = nbase + wn * 32 + nt * 16 + l15;
        float v = acc[mt][nt][r] + bias[gn];
        if (OUTMODE == 2) {
          out32[(size_t)gm * E_ + gn] = v;
        } else {
          const int bb = gm >> 11, s = gm & (S_ - 1);
          const int h  = gn >> 6,  d = gn & (D_ - 1);
          size_t idx = (OUTMODE == 0)
                           ? ((size_t)(bb * H_ + h) * S_ + s) * D_ + d
                           : ((size_t)(bb * H_ + h) * D_ + d) * S_ + s;
          out16[idx] = (_Float16)v;
        }
      }
    }
  }
}

// ---------------------------------------------------------------------------
// Fused attention: per block = (one bh, one 16-row q-tile):
//   Phase 1: scores = Q K^T / 8, mask -> f16 logits in LDS (full 2048 row).
//   Phase 2: row softmax in LDS (half8-vectorized); single f32 p_attn write.
//   Phase 3: Z = P V from LDS (contiguous f16 ds loads), split-K wave pairs.
// grid: (S/16, B*H), block 256 (8 waves).
// ---------------------------------------------------------------------------
#define SCPAD 2056  // 2048 + 8 pad: keeps rows 16B aligned, spreads banks

__global__ __launch_bounds__(256)
void attn_fused(const _Float16* __restrict__ Q16, const _Float16* __restrict__ K16,
                const _Float16* __restrict__ Vt16, const int* __restrict__ mask,
                float* __restrict__ pattn, _Float16* __restrict__ Z16) {
  __shared__ _Float16 sc[16][SCPAD];   // 65,792 B  (f16 logits / probs)
  __shared__ float zbuf[16][64];       //  4,096 B  (split-K partials)
  __shared__ float red[16][17];        //  1,088 B  (row reductions)

  const int tid  = threadIdx.x;
  const int lane = tid & 31, wave = tid >> 5;
  const int hl   = lane >> 4, l15 = lane & 15;
  const int bh   = blockIdx.y, b = bh >> 4, h = bh & 15;
  const int qbase = blockIdx.x * 16;

  const _Float16* Qh  = Q16 + (size_t)bh * S_ * D_;
  const _Float16* Kh  = K16 + (size_t)bh * S_ * D_;
  const _Float16* Vth = Vt16 + (size_t)bh * D_ * S_;

  const v8f zero = {0, 0, 0, 0, 0, 0, 0, 0};

  // ---- Phase 1: scores for all 2048 keys (wave w covers keys [w*256, +256))
  const _Float16* qr = Qh + (size_t)(qbase + l15) * D_;
  v16h af0 = load_frag(qr + hl * 8,      qr + 16 + hl * 8);
  v16h af1 = load_frag(qr + 32 + hl * 8, qr + 48 + hl * 8);
#pragma unroll 4
  for (int t = 0; t < 16; ++t) {
    const int nbase = wave * 256 + t * 16;
    const _Float16* kr = Kh + (size_t)(nbase + l15) * D_;
    if (t + 1 < 16) __builtin_prefetch(kr + 16 * D_, 0, 1);  // next K tile
    v8f acc = zero;
    acc = wmma_f16(af0, load_frag(kr + hl * 8, kr + 16 + hl * 8), acc);
    acc = wmma_f16(af1, load_frag(kr + 32 + hl * 8, kr + 48 + hl * 8), acc);
    const int ncol = nbase + l15;
    const int mk = mask[b * S_ + ncol];  // mask shape [B,1,1,S]
#pragma unroll
    for (int r = 0; r < 8; r++) {
      float v = acc[r] * 0.125f;         // 1/sqrt(64)
      if (mk == 0) v = -60000.0f;        // f16-representable; exp -> exactly 0
      sc[hl * 8 + r][ncol] = (_Float16)v;
    }
  }
  __syncthreads();

  // ---- Phase 2: softmax per row; 16 threads per row, 128 cols each,
  // all LDS traffic as half8 (16B) chunks, p_attn written as float4s.
  const int row = tid >> 4, sub = tid & 15;
  _Float16* srow = &sc[row][sub * 128];

  float lmax = -3.0e38f;
#pragma unroll
  for (int c = 0; c < 16; ++c) {
    half8 x = *(const half8*)(srow + c * 8);
#pragma unroll
    for (int i = 0; i < 8; ++i) lmax = fmaxf(lmax, (float)x[i]);
  }
  red[row][sub] = lmax;
  __syncthreads();
  if (sub == 0) {
    float m = red[row][0];
#pragma unroll
    for (int j = 1; j < 16; j++) m = fmaxf(m, red[row][j]);
    red[row][16] = m;
  }
  __syncthreads();
  const float m = red[row][16];

  float lsum = 0.f;
#pragma unroll
  for (int c = 0; c < 16; ++c) {
    half8 x = *(const half8*)(srow + c * 8);
    half8 e;
#pragma unroll
    for (int i = 0; i < 8; ++i) {
      float ev = __expf((float)x[i] - m);
      e[i] = (_Float16)ev;
      lsum += ev;
    }
    *(half8*)(srow + c * 8) = e;
  }
  red[row][sub] = lsum;
  __syncthreads();
  if (sub == 0) {
    float s = 0.f;
#pragma unroll
    for (int j = 0; j < 16; j++) s += red[row][j];
    red[row][16] = s;
  }
  __syncthreads();
  const float inv = 1.0f / red[row][16];

  float* grow = pattn + ((size_t)bh * S_ + qbase + row) * S_ + sub * 128;
#pragma unroll
  for (int c = 0; c < 16; ++c) {
    half8 e = *(const half8*)(srow + c * 8);
    half8 pn;
    f32x4 g0, g1;
#pragma unroll
    for (int i = 0; i < 4; ++i) {
      float p = (float)e[i] * inv;
      pn[i] = (_Float16)p;
      g0[i] = p;
    }
#pragma unroll
    for (int i = 0; i < 4; ++i) {
      float p = (float)e[4 + i] * inv;
      pn[4 + i] = (_Float16)p;
      g1[i] = p;
    }
    *(half8*)(srow + c * 8) = pn;          // normalized prob stays in LDS
    *(f32x4*)(grow + c * 8)     = g0;      // the single global p_attn write
    *(f32x4*)(grow + c * 8 + 4) = g1;
  }
  __syncthreads();

  // ---- Phase 3: Z(16x64) = P(16x2048) * V(2048x64), P from LDS.
  // wave -> (d-tile = wave&3, K-half = wave>>2); partials reduced via zbuf.
  const int dt = wave & 3;
  const int kh = wave >> 2;
  const int kbeg = kh * 1024;
  const _Float16* vrow = Vth + (size_t)(dt * 16 + l15) * S_;

  v8f zacc = zero;
  for (int kk = kbeg; kk < kbeg + 1024; kk += 32) {
    v16h a  = load_frag(&sc[l15][kk + hl * 8], &sc[l15][kk + 16 + hl * 8]);
    v16h bf = load_frag(vrow + kk + hl * 8, vrow + kk + 16 + hl * 8);
    zacc = wmma_f16(a, bf, zacc);
  }

  if (kh == 1) {
#pragma unroll
    for (int r = 0; r < 8; r++) zbuf[hl * 8 + r][dt * 16 + l15] = zacc[r];
  }
  __syncthreads();
  if (kh == 0) {
#pragma unroll
    for (int r = 0; r < 8; r++) {
      float v = zacc[r] + zbuf[hl * 8 + r][dt * 16 + l15];
      const int q = qbase + hl * 8 + r;
      Z16[((size_t)(b * S_ + q)) * E_ + h * D_ + dt * 16 + l15] = (_Float16)v;
    }
  }
}

// ---------------------------------------------------------------------------
// Host-side orchestration.
// ---------------------------------------------------------------------------
extern "C" void kernel_launch(void* const* d_in, const int* in_sizes, int n_in,
                              void* d_out, int out_size, void* d_ws, size_t ws_size,
                              hipStream_t stream) {
  (void)in_sizes; (void)n_in; (void)out_size; (void)ws_size;

  const float* queries = (const float*)d_in[0];
  const float* keys    = (const float*)d_in[1];
  const float* values  = (const float*)d_in[2];
  const int*   mask    = (const int*)d_in[3];
  const float* Wq = (const float*)d_in[4];
  const float* bq = (const float*)d_in[5];
  const float* Wk = (const float*)d_in[6];
  const float* bk = (const float*)d_in[7];
  const float* Wv = (const float*)d_in[8];
  const float* bv = (const float*)d_in[9];
  const float* Wo = (const float*)d_in[10];
  const float* bo = (const float*)d_in[11];

  float* out   = (float*)d_out;                 // [B,S,E]
  float* pattn = out + (size_t)B_ * S_ * E_;    // [B,H,S,S]

  char* ws = (char*)d_ws;
  const size_t MB = 1024 * 1024;
  _Float16* Xq16 = (_Float16*)(ws + 0 * MB);    // 8 MB each activation
  _Float16* Xk16 = (_Float16*)(ws + 8 * MB);
  _Float16* Xv16 = (_Float16*)(ws + 16 * MB);
  _Float16* Wq16 = (_Float16*)(ws + 24 * MB);   // 2 MB each weight
  _Float16* Wk16 = (_Float16*)(ws + 26 * MB);
  _Float16* Wv16 = (_Float16*)(ws + 28 * MB);
  _Float16* Wo16 = (_Float16*)(ws + 30 * MB);
  _Float16* Q16  = (_Float16*)(ws + 32 * MB);   // [b][h][s][d]
  _Float16* K16  = (_Float16*)(ws + 40 * MB);   // [b][h][s][d]
  _Float16* Vt16 = (_Float16*)(ws + 48 * MB);   // [b][h][d][s]
  _Float16* Z16  = (_Float16*)(ws + 56 * MB);   // [b*S][E]

  const int NA = B_ * S_ * E_;  // 4 Mi elements
  const int NW = E_ * E_;       // 1 Mi elements
  f32_to_f16_kernel<<<2048, 256, 0, stream>>>(queries, Xq16, NA);
  f32_to_f16_kernel<<<2048, 256, 0, stream>>>(keys,    Xk16, NA);
  f32_to_f16_kernel<<<2048, 256, 0, stream>>>(values,  Xv16, NA);
  f32_to_f16_kernel<<<1024, 256, 0, stream>>>(Wq, Wq16, NW);
  f32_to_f16_kernel<<<1024, 256, 0, stream>>>(Wk, Wk16, NW);
  f32_to_f16_kernel<<<1024, 256, 0, stream>>>(Wv, Wv16, NW);
  f32_to_f16_kernel<<<1024, 256, 0, stream>>>(Wo, Wo16, NW);

  dim3 gg(32, 8);  // 4096/128 x 1024/128
  gemm128<0><<<gg, 256, 0, stream>>>(Xq16, Wq16, bq, Q16,  nullptr);
  gemm128<0><<<gg, 256, 0, stream>>>(Xk16, Wk16, bk, K16,  nullptr);
  gemm128<1><<<gg, 256, 0, stream>>>(Xv16, Wv16, bv, Vt16, nullptr);

  attn_fused<<<dim3(S_ / 16, B_ * H_), 256, 0, stream>>>(Q16, K16, Vt16, mask,
                                                         pattn, Z16);

  gemm128<2><<<gg, 256, 0, stream>>>(Z16, Wo16, bo, nullptr, out);
}
